// Graph_PhysNet_85529978732658
// MI455X (gfx1250) — matrix operand order repaired and checked by reading, not compiled, not run on hardware
//
#include <hip/hip_runtime.h>

// ---------------------------------------------------------------------------
// PhysNet forward for MI455X (gfx1250, wave32, WMMA + TDM).
// GEMMs: f16 WMMA 16x16x32, f32 accumulate, 32 rows/wave, weights staged to
// LDS via tensor_load_to_lds (TDM). Pair path fused: rbf-filter GEMM +
// gather(xj) + multiply + sorted-segment scatter (run-length f32 atomics).
// ---------------------------------------------------------------------------

typedef _Float16 half_t;
typedef __attribute__((ext_vector_type(16))) _Float16 v16h;
typedef __attribute__((ext_vector_type(8)))  _Float16 v8h;
typedef __attribute__((ext_vector_type(8)))  float    v8f;
typedef __attribute__((ext_vector_type(4)))  unsigned int u32x4;
typedef __attribute__((ext_vector_type(8)))  int      i32x8;
typedef __attribute__((ext_vector_type(4)))  int      i32x4;

#define NATOMS 50000
#define NPAIRS 1000000
#define FDIM   128
#define KDIM   64
#define NBLK   5
#define NRI    3
#define NRF    2
#define NPAD   50048   // 391 * 128 (padded atom rows for unguarded A-frag loads)

#if defined(__has_builtin)
#if __has_builtin(__builtin_amdgcn_tensor_load_to_lds) && \
    __has_builtin(__builtin_amdgcn_s_wait_tensorcnt)
#define USE_TDM 1
#endif
#endif

__device__ __forceinline__ float ssp_f(float x) {
  // shifted softplus: log(0.5*exp(x)+0.5) = max(x,0) + log2(1+2^(-|x|*log2e))*ln2 - ln2
  // v_exp_f32 / v_log_f32 via amdgcn builtins (2^x and log2 x).
  const float LOG2E = 1.4426950408889634f;
  const float LN2   = 0.6931471805599453f;
  return fmaxf(x, 0.f) +
         LN2 * __builtin_amdgcn_logf(1.f + __builtin_amdgcn_exp2f(-fabsf(x) * LOG2E)) -
         LN2;
}

union AF { v16h v; v8h h[2]; };

__device__ __forceinline__ v8f wmma_f16(v16h a, v16h b, v8f c) {
  return __builtin_amdgcn_wmma_f32_16x16x32_f16(false, a, false, b,
                                                (short)0, c, false, false);
}

// A-fragment (16x32 f16): lane = row (lane&15), hsel = lane>>4.
// halves e<8  -> k = kbase + hsel*8 + e ; e>=8 -> k = kbase+16+hsel*8+(e-8)
__device__ __forceinline__ v16h load_a_frag(const half_t* rowp, int kbase, int hsel) {
  AF u;
  u.h[0] = *(const v8h*)(rowp + kbase + hsel * 8);
  u.h[1] = *(const v8h*)(rowp + kbase + 16 + hsel * 8);
  return u.v;
}

// Stage ndw contiguous dwords from global into LDS.
// TDM path: 1-D tensor descriptor (data_size=4B, tile_dim0=ndw<65536),
// issued by wave 0 only (EXEC ignored); fallback: cooperative b128 copy
// (128-thread workgroups).
__device__ __forceinline__ void stage_lds(const void* __restrict__ src,
                                          void* lds, int ndw) {
#ifdef USE_TDM
  if ((threadIdx.x >> 5) == 0) {
    const unsigned long long ga = (unsigned long long)(uintptr_t)src;
    const unsigned lds_addr = (unsigned)(uintptr_t)lds;   // low 32 bits = LDS offset
    const unsigned nd = (unsigned)ndw;
    u32x4 g0;
    g0[0] = 1u;                                  // count=1, user descriptor
    g0[1] = lds_addr;                            // D#.lds_addr
    g0[2] = (unsigned)(ga & 0xffffffffu);        // global_addr[31:0]
    g0[3] = (unsigned)((ga >> 32) & 0x01ffffffu) // global_addr[56:32]
            | (2u << 30);                        // type=2 ("image")
    i32x8 g1;
    g1[0] = (int)(2u << 16);                     // workgroup_mask=0, data_size=4B
    g1[1] = (int)((nd & 0xffffu) << 16);         // tensor_dim0 lo16
    g1[2] = (int)((nd >> 16) | (1u << 16));      // tensor_dim0 hi16, tensor_dim1=1
    g1[3] = (int)((nd & 0xffffu) << 16);         // tensor_dim1 hi16=0, tile_dim0=nd
    g1[4] = 0;                                   // tile_dim1=0 (unused), tile_dim2=0
    g1[5] = (int)nd;                             // tensor_dim0_stride lo32
    g1[6] = (int)((nd & 0xffffu) << 16);         // dim0_stride hi16=0, dim1_stride lo16
    g1[7] = 0;                                   // dim1_stride hi32
    i32x4 gz4 = {0, 0, 0, 0};
    i32x8 gz8 = {0, 0, 0, 0, 0, 0, 0, 0};
    __builtin_amdgcn_tensor_load_to_lds(g0, g1, gz4, gz4, gz8, 0);
    __builtin_amdgcn_s_wait_tensorcnt(0);
  }
#else
  {
    const uint4* s4 = (const uint4*)src;
    uint4* d4 = (uint4*)lds;
    for (int i = threadIdx.x; i < ndw / 4; i += 128) d4[i] = s4[i];
  }
#endif
  __syncthreads();
}

// ---------------------------------------------------------------------------
// Weight convert f32 -> f16 with transpose: dst[mat][col][row] = src[mat][row][col]
// ---------------------------------------------------------------------------
__global__ void wconv_kernel(const float* __restrict__ src, half_t* __restrict__ dst,
                             int rows, int cols, long total) {
  long t = (long)blockIdx.x * 256 + threadIdx.x;
  if (t >= total) return;
  int rc = rows * cols;
  long mat = t / rc;
  int rem = (int)(t - mat * rc);
  int rr = rem / cols, cc = rem - rr * cols;
  dst[mat * (long)rc + (long)cc * rows + rr] = (half_t)src[t];
}

// ---------------------------------------------------------------------------
// Elementwise: out_h[row][c] = (row < rows) ? f16(ssp(x[row][c])) : 0
// 8 elements / thread; zero-fills NPAD padding so GEMM A-loads are unguarded.
// ---------------------------------------------------------------------------
__global__ void ssp_h_kernel(const float* __restrict__ x, half_t* __restrict__ o,
                             int rows) {
  long t = ((long)blockIdx.x * 256 + threadIdx.x) * 8;
  if (t >= (long)NPAD * FDIM) return;
  int row = (int)(t >> 7);
  v8h r = {};
  if (row < rows) {
    v8f v = *(const v8f*)(x + t);
#pragma unroll
    for (int e = 0; e < 8; ++e) r[e] = (half_t)ssp_f(v[e]);
  }
  *(v8h*)(o + t) = r;
}

// ---------------------------------------------------------------------------
// GEMM: out = act( A[Mx128] @ W[128x128] + bias [+ uscale.*addend] )
// Compile-time variants (no runtime pointer checks in the epilogue):
//   ACT: apply ssp;  ADDEND/USCALE: fused residual / gated skip;
//   OUTH: f16 destination (zero-filled pad) else f32;  GUARD: row<M checks
//   (needed only when src/dst is an unpadded d_out slice).
// 4 waves x 32 rows = 128-row tile; K=128 in 4 steps; 64 WMMAs/wave; each
// B-fragment (ds_load_b128 x2) feeds two WMMAs.
// ---------------------------------------------------------------------------
template <bool ACT, bool ADDEND, bool USCALE, bool OUTH, bool GUARD>
__launch_bounds__(128)
__global__ void gemm128_kernel(const half_t* __restrict__ A,
                               const half_t* __restrict__ WT,
                               const float* __restrict__ bias,
                               const float* __restrict__ addend,
                               const float* __restrict__ uscale,
                               float* __restrict__ outf,
                               half_t* __restrict__ outh,
                               int M) {
  __shared__ half_t lw[FDIM * FDIM];
  stage_lds(WT, lw, FDIM * FDIM / 2);

  const int lane = threadIdx.x & 31;
  const int wave = threadIdx.x >> 5;
  const int lrow = lane & 15;
  const int hsel = lane >> 4;
  const int mbase = blockIdx.x * 128 + wave * 32;

  v16h af[2][4];
#pragma unroll
  for (int s = 0; s < 2; ++s) {
    const half_t* arow = A + (size_t)(mbase + s * 16 + lrow) * FDIM;
#pragma unroll
    for (int ks = 0; ks < 4; ++ks) af[s][ks] = load_a_frag(arow, ks * 32, hsel);
  }

#pragma unroll
  for (int nt = 0; nt < 8; ++nt) {
    const int nn = nt * 16 + lrow;
    v8f c0 = {}, c1 = {};
#pragma unroll
    for (int ks = 0; ks < 4; ++ks) {
      // B-fragment: lane = col nn, halves at k = ks*32 + hsel*8 and ks*32 + 16 + hsel*8
      AF b;
      const half_t* p = &lw[nn * FDIM + ks * 32 + hsel * 8];
      b.h[0] = *(const v8h*)(p);
      b.h[1] = *(const v8h*)(p + 16);
      c0 = wmma_f16(af[0][ks], b.v, c0);
      c1 = wmma_f16(af[1][ks], b.v, c1);
    }
    const float bv = bias[nn];
    float us = 1.f;
    if constexpr (USCALE) us = uscale[nn];
#pragma unroll
    for (int s = 0; s < 2; ++s) {
      const v8f& c = s ? c1 : c0;
#pragma unroll
      for (int r = 0; r < 8; ++r) {
        const int row = mbase + s * 16 + hsel * 8 + r;  // C layout: VGPR r -> M=8*hsel+r
        const size_t idx = (size_t)row * FDIM + nn;
        float v = c[r] + bv;
        if constexpr (ADDEND) {
          if (!GUARD || row < M) v += us * addend[idx];
        }
        if constexpr (ACT) v = ssp_f(v);
        if constexpr (OUTH) {
          outh[idx] = (half_t)(row < M ? v : 0.f);      // zero-fill pad rows
        } else {
          if (!GUARD || row < M) outf[idx] = v;
        }
      }
    }
  }
}

// ---------------------------------------------------------------------------
// Fused pair kernel. One wave per 16-pair tile (4 tiles / 128-thread WG):
//   g[16x128] = (cutoff .* rbf)[16x64]_f16 @ WgT   (WMMA, K=64 in 2 steps)
//   msg = g * xj[idx_j];  m[idx_i] += msg
// All 8 C-tiles computed first; scatter sweeps the 8 consecutive pairs a lane
// owns with ONE 64-bit base per row (8 loads / 8 atomics at immediate offsets
// nt*64B) and flushes at run boundaries of the sorted idx_i.
// ---------------------------------------------------------------------------
__launch_bounds__(128)
__global__ void pair_kernel(const float* __restrict__ cutoffs,
                            const float* __restrict__ rbfs,
                            const int* __restrict__ idx_i,
                            const int* __restrict__ idx_j,
                            const half_t* __restrict__ WgT,   // [128][64] f16
                            const float* __restrict__ xj,     // [NPAD][128] f32
                            float* __restrict__ m) {
  __shared__ half_t lwg[FDIM * KDIM];
  stage_lds(WgT, lwg, FDIM * KDIM / 2);

  const int lane = threadIdx.x & 31;
  const int wave = threadIdx.x >> 5;
  const int tile = blockIdx.x * 4 + wave;
  const int pbase = tile * 16;
  const int lrow = lane & 15;
  const int hsel = lane >> 4;

  // A-fragments: descriptor rows = cutoff * rbf, converted to f16
  const int p = pbase + lrow;
  const float cut = cutoffs[p];
  const float* rp = rbfs + (size_t)p * KDIM;
  v16h af[2];
#pragma unroll
  for (int ks = 0; ks < 2; ++ks) {
    v8f x0 = *(const v8f*)(rp + ks * 32 + hsel * 8);
    v8f x1 = *(const v8f*)(rp + ks * 32 + 16 + hsel * 8);
    AF u;
#pragma unroll
    for (int e = 0; e < 8; ++e) {
      u.v[e]     = (half_t)(cut * x0[e]);
      u.v[e + 8] = (half_t)(cut * x1[e]);
    }
    af[ks] = u.v;
  }

  // The 8 consecutive pair indices this lane's C-rows correspond to
  int ii[8], jj[8];
#pragma unroll
  for (int r = 0; r < 8; ++r) {
    const int pp = pbase + 8 * hsel + r;
    ii[r] = idx_i[pp];
    jj[r] = idx_j[pp];
  }

  // All 8 accumulator tiles: g columns n = nt*16 + lrow
  v8f acc[8];
#pragma unroll
  for (int nt = 0; nt < 8; ++nt) {
    v8f c = {};
#pragma unroll
    for (int ks = 0; ks < 2; ++ks) {
      AF b;
      const half_t* wp = &lwg[(nt * 16 + lrow) * KDIM + ks * 32 + hsel * 8];
      b.h[0] = *(const v8h*)(wp);
      b.h[1] = *(const v8h*)(wp + 16);
      c = wmma_f16(af[ks], b.v, c);
    }
    acc[nt] = c;
  }

  // Gather + multiply + run-length-compressed scatter (idx_i sorted).
  float sum[8] = {0.f, 0.f, 0.f, 0.f, 0.f, 0.f, 0.f, 0.f};
#pragma unroll
  for (int r = 0; r < 8; ++r) {
    const float* xjr = xj + (size_t)jj[r] * FDIM + lrow;   // one base, 8 imm offsets
#pragma unroll
    for (int nt = 0; nt < 8; ++nt) sum[nt] += acc[nt][r] * xjr[nt * 16];
    if ((r == 7) || (ii[r + 1] != ii[r])) {                // flush at run boundary
      float* mr = m + (size_t)ii[r] * FDIM + lrow;         // one base, 8 imm offsets
#pragma unroll
      for (int nt = 0; nt < 8; ++nt) {
        __hip_atomic_fetch_add(mr + nt * 16, sum[nt],
                               __ATOMIC_RELAXED, __HIP_MEMORY_SCOPE_AGENT);
        sum[nt] = 0.f;
      }
    }
  }
}

// ---------------------------------------------------------------------------
// Host driver
// ---------------------------------------------------------------------------
extern "C" void kernel_launch(void* const* d_in, const int* in_sizes, int n_in,
                              void* d_out, int out_size, void* d_ws, size_t ws_size,
                              hipStream_t stream) {
  const float* features = (const float*)d_in[0];
  const float* cutoffs  = (const float*)d_in[1];
  const float* rbfs     = (const float*)d_in[2];
  const int*   idx_i    = (const int*)d_in[3];
  const int*   idx_j    = (const int*)d_in[4];
  const float* Wg   = (const float*)d_in[5];
  const float* Wi   = (const float*)d_in[6];
  const float* bi   = (const float*)d_in[7];
  const float* Wj   = (const float*)d_in[8];
  const float* bj   = (const float*)d_in[9];
  const float* Wr1  = (const float*)d_in[10];
  const float* br1  = (const float*)d_in[11];
  const float* Wr2  = (const float*)d_in[12];
  const float* br2  = (const float*)d_in[13];
  const float* Wout = (const float*)d_in[14];
  const float* bout = (const float*)d_in[15];
  const float* u    = (const float*)d_in[16];
  const float* Wf1  = (const float*)d_in[17];
  const float* bf1  = (const float*)d_in[18];
  const float* Wf2  = (const float*)d_in[19];
  const float* bf2  = (const float*)d_in[20];
  float* out = (float*)d_out;
  (void)in_sizes; (void)n_in; (void)out_size; (void)ws_size;

  // Workspace carve-out (256B aligned)
  char* ws = (char*)d_ws;
  size_t off = 0;
  auto carve = [&](size_t bytes) -> char* {
    char* pp = ws + off;
    off = (off + bytes + 255) & ~(size_t)255;
    return pp;
  };
  half_t* WgT   = (half_t*)carve((size_t)NBLK * KDIM * FDIM * 2);
  half_t* WiT   = (half_t*)carve((size_t)NBLK * FDIM * FDIM * 2);
  half_t* WjT   = (half_t*)carve((size_t)NBLK * FDIM * FDIM * 2);
  half_t* WoutT = (half_t*)carve((size_t)NBLK * FDIM * FDIM * 2);
  half_t* Wr1T  = (half_t*)carve((size_t)NBLK * NRI * FDIM * FDIM * 2);
  half_t* Wr2T  = (half_t*)carve((size_t)NBLK * NRI * FDIM * FDIM * 2);
  half_t* Wf1T  = (half_t*)carve((size_t)NBLK * NRF * FDIM * FDIM * 2);
  half_t* Wf2T  = (half_t*)carve((size_t)NBLK * NRF * FDIM * FDIM * 2);
  half_t* xa_h  = (half_t*)carve((size_t)NPAD * FDIM * 2);   // ssp(x)/ssp(m) staging
  half_t* t2_h  = (half_t*)carve((size_t)NPAD * FDIM * 2);
  float*  xjbuf = (float*) carve((size_t)NPAD * FDIM * 4);
  float*  mbuf  = (float*) carve((size_t)NPAD * FDIM * 4);

  // Convert + transpose weights to f16 (tiny)
  auto wc = [&](const float* s, half_t* d, int rows, int cols, int nmat) {
    long total = (long)nmat * rows * cols;
    wconv_kernel<<<dim3((unsigned)((total + 255) / 256)), dim3(256), 0, stream>>>(
        s, d, rows, cols, total);
  };
  wc(Wg, WgT, KDIM, FDIM, NBLK);
  wc(Wi, WiT, FDIM, FDIM, NBLK);
  wc(Wj, WjT, FDIM, FDIM, NBLK);
  wc(Wout, WoutT, FDIM, FDIM, NBLK);
  wc(Wr1, Wr1T, FDIM, FDIM, NBLK * NRI);
  wc(Wr2, Wr2T, FDIM, FDIM, NBLK * NRI);
  wc(Wf1, Wf1T, FDIM, FDIM, NBLK * NRF);
  wc(Wf2, Wf2T, FDIM, FDIM, NBLK * NRF);

  const dim3 gG(NPAD / 128), bG(128);                        // 391 x 128
  const unsigned sspG = (unsigned)((size_t)NPAD * FDIM / 8 / 256);  // 3128
  const dim3 gP(NPAIRS / 16 / 4), bP(128);                   // 15625 x 128

  for (int b = 0; b < NBLK; ++b) {
    const float* xprev = (b == 0) ? features : out + (size_t)(b - 1) * NATOMS * FDIM;
    float* xcur = out + (size_t)b * NATOMS * FDIM;

    // xa = ssp(x)  (f16, padded)
    ssp_h_kernel<<<sspG, 256, 0, stream>>>(xprev, xa_h, NATOMS);
    // m = xi = ssp(xa @ Wi + bi)   [padded f32 target: no guard]
    gemm128_kernel<true, false, false, false, false><<<gG, bG, 0, stream>>>(
        xa_h, WiT + (size_t)b * FDIM * FDIM, bi + b * FDIM,
        nullptr, nullptr, mbuf, nullptr, NATOMS);
    // xj = ssp(xa @ Wj + bj)
    gemm128_kernel<true, false, false, false, false><<<gG, bG, 0, stream>>>(
        xa_h, WjT + (size_t)b * FDIM * FDIM, bj + b * FDIM,
        nullptr, nullptr, xjbuf, nullptr, NATOMS);
    // m += segment_sum(g * xj[idx_j], idx_i)
    pair_kernel<<<gP, bP, 0, stream>>>(cutoffs, rbfs, idx_i, idx_j,
                                       WgT + (size_t)b * KDIM * FDIM, xjbuf, mbuf);
    // interaction residuals: m = m + ssp(ssp(m)@Wr1+br1)@Wr2 + br2
    for (int r = 0; r < NRI; ++r) {
      const size_t wi = (size_t)(b * NRI + r) * FDIM * FDIM;
      const int bo = (b * NRI + r) * FDIM;
      ssp_h_kernel<<<sspG, 256, 0, stream>>>(mbuf, xa_h, NATOMS);
      gemm128_kernel<true, false, false, true, false><<<gG, bG, 0, stream>>>(
          xa_h, Wr1T + wi, br1 + bo, nullptr, nullptr, nullptr, t2_h, NATOMS);
      gemm128_kernel<false, true, false, false, false><<<gG, bG, 0, stream>>>(
          t2_h, Wr2T + wi, br2 + bo, mbuf, nullptr, mbuf, nullptr, NATOMS);
    }
    // x = u*x + ssp(m) @ Wout + bout   [d_out slice: guarded]
    ssp_h_kernel<<<sspG, 256, 0, stream>>>(mbuf, xa_h, NATOMS);
    gemm128_kernel<false, true, true, false, true><<<gG, bG, 0, stream>>>(
        xa_h, WoutT + (size_t)b * FDIM * FDIM, bout + b * FDIM,
        xprev, u + b * FDIM, xcur, nullptr, NATOMS);
    // feature residuals: x = x + ssp(ssp(x)@Wf1+bf1)@Wf2 + bf2
    for (int r = 0; r < NRF; ++r) {
      const size_t wi = (size_t)(b * NRF + r) * FDIM * FDIM;
      const int bo = (b * NRF + r) * FDIM;
      ssp_h_kernel<<<sspG, 256, 0, stream>>>(xcur, xa_h, NATOMS);
      gemm128_kernel<true, false, false, true, false><<<gG, bG, 0, stream>>>(
          xa_h, Wf1T + wi, bf1 + bo, nullptr, nullptr, nullptr, t2_h, NATOMS);
      gemm128_kernel<false, true, false, false, true><<<gG, bG, 0, stream>>>(
          t2_h, Wf2T + wi, bf2 + bo, xcur, nullptr, xcur, nullptr, NATOMS);
    }
  }
}
